// DeepGATEncoder_38981123178586
// MI455X (gfx1250) — matrix operand order repaired and checked by reading, compiled, not run on hardware
//
#include <hip/hip_runtime.h>
#include <hip/hip_bf16.h>
#include <cstring>

typedef __attribute__((ext_vector_type(16))) __bf16 v16bf;
typedef __attribute__((ext_vector_type(8)))  float  v8f;
typedef unsigned int u32x4 __attribute__((ext_vector_type(4)));
typedef int          i32x8 __attribute__((ext_vector_type(8)));
typedef int          i32x4 __attribute__((ext_vector_type(4)));

#define Bc    8
#define Nn    2048
#define Ee    32768
#define HID   64
#define LAT   32
#define H1c   4
#define H2c   2
#define K1c   1024
#define K2c   205
#define D1c   256
#define D2c   128
#define BN1   (Bc*Nn)        // 16384
#define BK1   (Bc*K1c)       // 8192
#define M1    (Bc*Ee + BN1)  // 278528 edges + self loops
#define M2    (Bc*Ee + BK1)  // 270336

__host__ __device__ inline unsigned enc_f(float f) {
  unsigned u;
#ifdef __HIP_DEVICE_COMPILE__
  u = __float_as_uint(f);
#else
  memcpy(&u, &f, 4);
#endif
  return (u & 0x80000000u) ? ~u : (u | 0x80000000u);
}
__device__ inline float dec_f(unsigned u) {
  return __uint_as_float((u & 0x80000000u) ? (u & 0x7FFFFFFFu) : ~u);
}

// ---------------------------------------------------------------- utilities
__global__ void fill_u32(unsigned* p, unsigned v, int n) {
  int i = blockIdx.x * blockDim.x + threadIdx.x;
  if (i < n) p[i] = v;
}

__global__ void norms_kernel(const float* pw1, const float* pw2, float* norms) {
  __shared__ float red[256];
  int t = threadIdx.x;
  red[t] = (t < D1c) ? pw1[t] * pw1[t] : 0.f;
  __syncthreads();
  for (int s = 128; s > 0; s >>= 1) { if (t < s) red[t] += red[t + s]; __syncthreads(); }
  if (t == 0) norms[0] = sqrtf(red[0]);
  __syncthreads();
  red[t] = (t < D2c) ? pw2[t] * pw2[t] : 0.f;
  __syncthreads();
  for (int s = 128; s > 0; s >>= 1) { if (t < s) red[t] += red[t + s]; __syncthreads(); }
  if (t == 0) norms[1] = sqrtf(red[0]);
}

// ------------------------------------------------- stage-1 linear (K = 2)
__global__ void lin1_kernel(const float* __restrict__ x,
                            const float* __restrict__ Wl, const float* __restrict__ bl,
                            const float* __restrict__ Wr, const float* __restrict__ br,
                            float* __restrict__ xl, float* __restrict__ xr) {
  int i = blockIdx.x * blockDim.x + threadIdx.x;
  if (i >= BN1 * D1c) return;
  int n = i / D1c, j = i - n * D1c;
  float x0 = x[n * 2 + 0], x1 = x[n * 2 + 1];
  xl[i] = x0 * Wl[j] + x1 * Wl[D1c + j] + bl[j];
  xr[i] = x0 * Wr[j] + x1 * Wr[D1c + j] + br[j];
}

// --------------------------------------------- generic GATv2 edge kernels
// one wave of 32 lanes per edge; edges [0,B*E) real, rest self-loops
__global__ void edge_score(const float* __restrict__ xl, const float* __restrict__ xr,
                           const float* __restrict__ att, const int* __restrict__ src,
                           const int* __restrict__ dst, const int* __restrict__ valid,
                           float* __restrict__ e_ws, unsigned* __restrict__ mx,
                           int nPer, int H, int M) {
  int m = blockIdx.x * 8 + (threadIdx.x >> 5);
  if (m >= M) return;
  int lane = threadIdx.x & 31;
  int gs, gd; bool ok;
  if (m < Bc * Ee) {
    ok = (!valid) || (valid[m] != 0);
    int b = m / Ee;
    gs = b * nPer + src[m];
    gd = b * nPer + dst[m];
  } else { gs = gd = m - Bc * Ee; ok = true; }
  if (!ok) return;
  int D = H * HID;
  const float* pj = xl + (size_t)gs * D;
  const float* pi = xr + (size_t)gd * D;
  for (int h = 0; h < H; ++h) {
    float a = 0.f;
    for (int c = lane; c < HID; c += 32) {
      int j = h * HID + c;
      float v = pj[j] + pi[j];
      v = v > 0.f ? v : 0.2f * v;          // leaky_relu(0.2)
      a += att[j] * v;
    }
    for (int off = 16; off > 0; off >>= 1) a += __shfl_xor(a, off, 32);
    if (lane == 0) {
      e_ws[(size_t)m * H + h] = a;
      atomicMax(mx + (size_t)gd * H + h, enc_f(a));   // segment max
    }
  }
}

__global__ void edge_denom(const int* __restrict__ dst, const int* __restrict__ valid,
                           const unsigned* __restrict__ mx, float* __restrict__ e_ws,
                           float* __restrict__ den, int nPer, int H, int M) {
  int m = blockIdx.x * blockDim.x + threadIdx.x;
  if (m >= M) return;
  int gd; bool ok;
  if (m < Bc * Ee) {
    ok = (!valid) || (valid[m] != 0);
    int b = m / Ee;
    gd = b * nPer + dst[m];
  } else { gd = m - Bc * Ee; ok = true; }
  if (!ok) return;
  for (int h = 0; h < H; ++h) {
    float mv = dec_f(mx[(size_t)gd * H + h]);
    float p = __expf(e_ws[(size_t)m * H + h] - mv);
    e_ws[(size_t)m * H + h] = p;
    atomicAdd(den + (size_t)gd * H + h, p);
  }
}

__global__ void edge_agg(const float* __restrict__ xl, const int* __restrict__ src,
                         const int* __restrict__ dst, const int* __restrict__ valid,
                         const float* __restrict__ e_ws, const float* __restrict__ den,
                         float* __restrict__ agg, int nPer, int H, int M) {
  int m = blockIdx.x * 8 + (threadIdx.x >> 5);
  if (m >= M) return;
  int lane = threadIdx.x & 31;
  int gs, gd; bool ok;
  if (m < Bc * Ee) {
    ok = (!valid) || (valid[m] != 0);
    int b = m / Ee;
    gs = b * nPer + src[m];
    gd = b * nPer + dst[m];
  } else { gs = gd = m - Bc * Ee; ok = true; }
  if (!ok) return;
  const float* pj = xl + (size_t)gs * (H * HID);
  float* po = agg + (size_t)gd * (H * HID);
  for (int h = 0; h < H; ++h) {
    float alpha = e_ws[(size_t)m * H + h] / (den[(size_t)gd * H + h] + 1e-16f);
    for (int c = lane; c < HID; c += 32)
      atomicAdd(po + h * HID + c, alpha * pj[h * HID + c]);
  }
}

// ---------------------------------------- bias + BN(eval) + ELU + pool score
__global__ void bn_elu_score(const float* __restrict__ agg, const float* __restrict__ bias,
                             const float* __restrict__ g, const float* __restrict__ be,
                             const float* __restrict__ mu, const float* __restrict__ var,
                             const float* __restrict__ pw, const float* __restrict__ norm,
                             float* __restrict__ hout, float* __restrict__ score,
                             int D, int NN) {
  int nidx = blockIdx.x * 8 + (threadIdx.x >> 5);
  if (nidx >= NN) return;
  int lane = threadIdx.x & 31;
  float s = 0.f;
  const float* pa = agg + (size_t)nidx * D;
  float* ph = hout + (size_t)nidx * D;
  for (int j = lane; j < D; j += 32) {
    float v = pa[j] + bias[j];
    v = (v - mu[j]) * g[j] * rsqrtf(var[j] + 1e-5f) + be[j];
    v = v > 0.f ? v : (__expf(v) - 1.f);            // ELU
    ph[j] = v;
    s += v * pw[j];
  }
  for (int off = 16; off > 0; off >>= 1) s += __shfl_xor(s, off, 32);
  if (lane == 0) score[nidx] = s / (norm[0] + 1e-16f);
}

// ------------------------------------- per-graph top-k via LDS bitonic sort
__global__ void topk_kernel(const float* __restrict__ score, int n, int k,
                            int* __restrict__ idx_out, float* __restrict__ val_out,
                            int* __restrict__ mapping) {
  __shared__ float ssc[2048];
  __shared__ int   sid[2048];
  int b = blockIdx.x;
  const float* sc = score + (size_t)b * n;
  for (int i = threadIdx.x; i < n; i += blockDim.x) { ssc[i] = sc[i]; sid[i] = i; }
  if (mapping)
    for (int i = threadIdx.x; i < n; i += blockDim.x) mapping[(size_t)b * n + i] = -1;
  __syncthreads();
  for (int ksz = 2; ksz <= n; ksz <<= 1) {
    for (int j = ksz >> 1; j > 0; j >>= 1) {
      for (int t = threadIdx.x; t < (n >> 1); t += blockDim.x) {
        int i = 2 * t - (t & (j - 1));
        int ixj = i + j;
        bool desc = ((i & ksz) == 0);
        float s0 = ssc[i], s1 = ssc[ixj];
        int i0 = sid[i], i1 = sid[ixj];
        bool firstBigger = (s0 > s1) || (s0 == s1 && i0 < i1);
        if (desc != firstBigger) {
          ssc[i] = s1; ssc[ixj] = s0; sid[i] = i1; sid[ixj] = i0;
        }
      }
      __syncthreads();
    }
  }
  for (int t = threadIdx.x; t < k; t += blockDim.x) {
    idx_out[(size_t)b * k + t] = sid[t];
    val_out[(size_t)b * k + t] = ssc[t];
    if (mapping) mapping[(size_t)b * n + sid[t]] = t;
  }
}

// ---------------------------- pool-1 gather + tanh gate, emit bf16 GEMM input
__global__ void gather_pool1(const float* __restrict__ h1, const int* __restrict__ idx,
                             const float* __restrict__ vals, __bf16* __restrict__ xkbf) {
  int r = blockIdx.x * 8 + (threadIdx.x >> 5);
  if (r >= BK1) return;
  int lane = threadIdx.x & 31;
  int b = r / K1c;
  int node = b * Nn + idx[r];
  float gate = tanhf(vals[r]);
  const float* ph = h1 + (size_t)node * D1c;
  __bf16* po = xkbf + (size_t)r * D1c;
  for (int j = lane; j < D1c; j += 32) po[j] = (__bf16)(ph[j] * gate);
}

__global__ void relabel1(const int* __restrict__ map1, const int* __restrict__ src,
                         const int* __restrict__ dst, int* __restrict__ s1e,
                         int* __restrict__ d1e, int* __restrict__ v1e) {
  int i = blockIdx.x * blockDim.x + threadIdx.x;
  if (i >= Bc * Ee) return;
  int b = i / Ee;
  int ns = map1[b * Nn + src[i]];
  int nd = map1[b * Nn + dst[i]];
  v1e[i] = (ns >= 0 && nd >= 0) ? 1 : 0;
  s1e[i] = ns < 0 ? 0 : ns;
  d1e[i] = nd < 0 ? 0 : nd;
}

// weight [K,Nc] f32 -> transposed [Nc,K] bf16 (contiguous B-fragment loads)
__global__ void wT_bf16(const float* __restrict__ W, __bf16* __restrict__ WT,
                        int K, int Nc) {
  int i = blockIdx.x * blockDim.x + threadIdx.x;
  if (i >= K * Nc) return;
  int n = i / K, kk = i - n * K;
  WT[i] = (__bf16)W[(size_t)kk * Nc + n];
}

// --------------------- WMMA bf16 GEMM: C[M,Nc] = A[M,K] @ W[K,Nc] + bias[Nc]
// All 4 waves in a block share the same 16 output columns, so the 16xK bf16
// weight tile (8 KB) is staged into LDS ONCE per block by the Tensor Data
// Mover (2-D descriptor: tile_dim0 = K contiguous elems, tile_dim1 = 16 rows,
// stride = K, data_size = 2 B), completion via s_wait_tensorcnt + barrier.
// B fragments are then ds_load'ed; A fragments stream from global.
__global__ void wmma_gemm_bf16(const __bf16* __restrict__ A, const __bf16* __restrict__ WT,
                               const float* __restrict__ bias, float* __restrict__ C,
                               int Mrows, int K, int Nc) {
  __shared__ __bf16 ldsB[16 * D1c];            // 16 columns x K (K <= 256)
  int lane = threadIdx.x & 31;
  int wave = threadIdx.x >> 5;
  int nbase = blockIdx.x * 16;

  if (wave == 0) {                             // one TDM issue per block
    unsigned long long ga =
        (unsigned long long)(const void*)(WT + (size_t)nbase * K);
    unsigned lds_off = (unsigned)(unsigned long long)(const void*)(&ldsB[0]);
    u32x4 g0;
    g0[0] = 1u;                                   // count=1 (valid user D#)
    g0[1] = lds_off;                              // lds_addr      [63:32]
    g0[2] = (unsigned)(ga & 0xFFFFFFFFu);         // global_addr   [95:64]
    g0[3] = (unsigned)((ga >> 32) & 0x01FFFFFFu)  // global_addr   [120:96]
            | 0x80000000u;                        // type=2 (image) [127:126]
    i32x8 g1;
    g1[0] = 0x00010000;                           // data_size=1 -> 2 bytes
    g1[1] = (int)(((unsigned)K & 0xFFFFu) << 16); // tensor_dim0[15:0]
    g1[2] = (int)((((unsigned)K >> 16) & 0xFFFFu) // tensor_dim0[31:16]
                  | (16u << 16));                 // tensor_dim1[15:0] = 16
    g1[3] = (int)(((unsigned)K & 0xFFFFu) << 16); // tile_dim0 = K  [127:112]
    g1[4] = 16;                                   // tile_dim1 = 16 rows
    g1[5] = K;                                    // tensor_dim0_stride lo32
    g1[6] = 0;                                    // stride hi / dim1_stride lo
    g1[7] = 0;
    i32x4 gz4 = {0, 0, 0, 0};                     // 2-D tensor: groups 2/3 null
    i32x8 gz8 = {0, 0, 0, 0, 0, 0, 0, 0};
    __builtin_amdgcn_tensor_load_to_lds(g0, g1, gz4, gz4, gz8, 0);
    __builtin_amdgcn_s_wait_tensorcnt(0);
  }
  __syncthreads();

  int mbase = (blockIdx.y * 4 + wave) * 16;
  if (mbase >= Mrows) return;                  // wave-uniform: EXEC stays all-1
  int hi = lane >> 4;                          // K-half select
  int col = nbase + (lane & 15);
  int arow = mbase + (lane & 15);
  float bv = bias[col];
  v8f c;
#pragma unroll
  for (int r = 0; r < 8; ++r) c[r] = bv;
  for (int k0 = 0; k0 < K; k0 += 32) {
    v16bf a, bfr;
    const __bf16* ap = A + (size_t)arow * K + k0 + hi * 8;
#pragma unroll
    for (int i = 0; i < 8; ++i) { a[i] = ap[i]; a[8 + i] = ap[16 + i]; }
    const __bf16* bp = &ldsB[(lane & 15) * K + k0 + hi * 8];
#pragma unroll
    for (int i = 0; i < 8; ++i) { bfr[i] = bp[i]; bfr[8 + i] = bp[16 + i]; }
    c = __builtin_amdgcn_wmma_f32_16x16x32_bf16(false, a, false, bfr,
                                                (short)0, c, false, false);
  }
#pragma unroll
  for (int r = 0; r < 8; ++r)
    C[(size_t)(mbase + hi * 8 + r) * Nc + col] = c[r];
}

// ---------------------------------- pool-2 gate + global mean over K2 nodes
__global__ void pool_mean(const float* __restrict__ h2, const int* __restrict__ idx2,
                          const float* __restrict__ val2, float* __restrict__ pooled) {
  int b = blockIdx.x;
  int j = threadIdx.x;                       // blockDim = D2
  float s = 0.f;
  for (int t = 0; t < K2c; ++t) {
    int node = idx2[b * K2c + t];
    float g = tanhf(val2[b * K2c + t]);
    s += h2[(size_t)(b * K1c + node) * D2c + j] * g;
  }
  pooled[b * D2c + j] = s / (float)K2c;
}

__global__ void mlp_head(const float* __restrict__ pooled, const float* __restrict__ W3,
                         const float* __restrict__ b3, const float* __restrict__ W4,
                         const float* __restrict__ b4, float* __restrict__ out) {
  __shared__ float hid[Bc][HID];
  for (int i = threadIdx.x; i < Bc * HID; i += blockDim.x) {
    int b = i / HID, k = i - b * HID;
    float s = b3[k];
    for (int j = 0; j < D2c; ++j) s += pooled[b * D2c + j] * W3[j * HID + k];
    hid[b][k] = fmaxf(s, 0.f);
  }
  __syncthreads();
  for (int i = threadIdx.x; i < Bc * LAT; i += blockDim.x) {
    int b = i / LAT, l = i - b * LAT;
    float s = b4[l];
    for (int k = 0; k < HID; ++k) s += hid[b][k] * W4[k * LAT + l];
    out[i] = s;
  }
}

// ============================================================== launcher
extern "C" void kernel_launch(void* const* d_in, const int* in_sizes, int n_in,
                              void* d_out, int out_size, void* d_ws, size_t ws_size,
                              hipStream_t stream) {
  (void)in_sizes; (void)n_in; (void)out_size; (void)ws_size;
  const float* x    = (const float*)d_in[0];
  const int*   srcI = (const int*)d_in[1];
  const int*   dstI = (const int*)d_in[2];
  const float* Wl1  = (const float*)d_in[3];  const float* bl1 = (const float*)d_in[4];
  const float* Wr1  = (const float*)d_in[5];  const float* br1 = (const float*)d_in[6];
  const float* att1 = (const float*)d_in[7];  const float* bias1 = (const float*)d_in[8];
  const float* g1   = (const float*)d_in[9];  const float* be1 = (const float*)d_in[10];
  const float* mu1  = (const float*)d_in[11]; const float* var1 = (const float*)d_in[12];
  const float* pw1  = (const float*)d_in[13];
  const float* Wl2  = (const float*)d_in[14]; const float* bl2 = (const float*)d_in[15];
  const float* Wr2  = (const float*)d_in[16]; const float* br2 = (const float*)d_in[17];
  const float* att2 = (const float*)d_in[18]; const float* bias2 = (const float*)d_in[19];
  const float* g2   = (const float*)d_in[20]; const float* be2 = (const float*)d_in[21];
  const float* mu2  = (const float*)d_in[22]; const float* var2 = (const float*)d_in[23];
  const float* pw2  = (const float*)d_in[24];
  const float* W3   = (const float*)d_in[25]; const float* b3 = (const float*)d_in[26];
  const float* W4   = (const float*)d_in[27]; const float* b4 = (const float*)d_in[28];

  char* w = (char*)d_ws; size_t off = 0;
  auto alloc = [&](size_t bytes) -> void* {
    off = (off + 255) & ~(size_t)255;
    void* p = w + off; off += bytes; return p;
  };
  float*    xl1   = (float*)alloc((size_t)BN1 * D1c * 4);
  float*    xr1   = (float*)alloc((size_t)BN1 * D1c * 4);
  float*    e1    = (float*)alloc((size_t)M1 * H1c * 4);
  unsigned* max1  = (unsigned*)alloc((size_t)BN1 * H1c * 4);
  float*    den1  = (float*)alloc((size_t)BN1 * H1c * 4);
  float*    agg1  = (float*)alloc((size_t)BN1 * D1c * 4);
  float*    h1    = (float*)alloc((size_t)BN1 * D1c * 4);
  float*    sc1   = (float*)alloc((size_t)BN1 * 4);
  int*      idx1  = (int*)alloc((size_t)BK1 * 4);
  float*    val1  = (float*)alloc((size_t)BK1 * 4);
  int*      map1  = (int*)alloc((size_t)BN1 * 4);
  __bf16*   xkbf  = (__bf16*)alloc((size_t)BK1 * D1c * 2);
  __bf16*   Wl2T  = (__bf16*)alloc((size_t)D1c * D2c * 2);
  __bf16*   Wr2T  = (__bf16*)alloc((size_t)D1c * D2c * 2);
  float*    xl2   = (float*)alloc((size_t)BK1 * D2c * 4);
  float*    xr2   = (float*)alloc((size_t)BK1 * D2c * 4);
  int*      s1e   = (int*)alloc((size_t)Bc * Ee * 4);
  int*      d1e   = (int*)alloc((size_t)Bc * Ee * 4);
  int*      v1e   = (int*)alloc((size_t)Bc * Ee * 4);
  float*    e2    = (float*)alloc((size_t)M2 * H2c * 4);
  unsigned* max2  = (unsigned*)alloc((size_t)BK1 * H2c * 4);
  float*    den2  = (float*)alloc((size_t)BK1 * H2c * 4);
  float*    agg2  = (float*)alloc((size_t)BK1 * D2c * 4);
  float*    h2    = (float*)alloc((size_t)BK1 * D2c * 4);
  float*    sc2   = (float*)alloc((size_t)BK1 * 4);
  int*      idx2  = (int*)alloc((size_t)Bc * K2c * 4);
  float*    val2  = (float*)alloc((size_t)Bc * K2c * 4);
  float*    pooled = (float*)alloc((size_t)Bc * D2c * 4);
  float*    norms = (float*)alloc(2 * 4);

  const unsigned negInf = enc_f(-3e38f);
  auto grid1 = [](int n, int b) { return (n + b - 1) / b; };

  norms_kernel<<<1, 256, 0, stream>>>(pw1, pw2, norms);
  fill_u32<<<grid1(BN1 * H1c, 256), 256, 0, stream>>>(max1, negInf, BN1 * H1c);
  fill_u32<<<grid1(BN1 * H1c, 256), 256, 0, stream>>>((unsigned*)den1, 0u, BN1 * H1c);
  fill_u32<<<grid1(BN1 * D1c, 256), 256, 0, stream>>>((unsigned*)agg1, 0u, BN1 * D1c);
  fill_u32<<<grid1(BK1 * H2c, 256), 256, 0, stream>>>(max2, negInf, BK1 * H2c);
  fill_u32<<<grid1(BK1 * H2c, 256), 256, 0, stream>>>((unsigned*)den2, 0u, BK1 * H2c);
  fill_u32<<<grid1(BK1 * D2c, 256), 256, 0, stream>>>((unsigned*)agg2, 0u, BK1 * D2c);

  // ---- GAT layer 1
  lin1_kernel<<<grid1(BN1 * D1c, 256), 256, 0, stream>>>(x, Wl1, bl1, Wr1, br1, xl1, xr1);
  edge_score<<<grid1(M1, 8), 256, 0, stream>>>(xl1, xr1, att1, srcI, dstI, nullptr,
                                               e1, max1, Nn, H1c, M1);
  edge_denom<<<grid1(M1, 256), 256, 0, stream>>>(dstI, nullptr, max1, e1, den1, Nn, H1c, M1);
  edge_agg<<<grid1(M1, 8), 256, 0, stream>>>(xl1, srcI, dstI, nullptr, e1, den1,
                                             agg1, Nn, H1c, M1);
  bn_elu_score<<<grid1(BN1, 8), 256, 0, stream>>>(agg1, bias1, g1, be1, mu1, var1,
                                                  pw1, norms + 0, h1, sc1, D1c, BN1);
  // ---- pool 1
  topk_kernel<<<Bc, 512, 0, stream>>>(sc1, Nn, K1c, idx1, val1, map1);
  gather_pool1<<<grid1(BK1, 8), 256, 0, stream>>>(h1, idx1, val1, xkbf);
  relabel1<<<grid1(Bc * Ee, 256), 256, 0, stream>>>(map1, srcI, dstI, s1e, d1e, v1e);

  // ---- GAT layer 2 linears via TDM-staged LDS + WMMA (bf16 in, f32 accum)
  wT_bf16<<<grid1(D1c * D2c, 256), 256, 0, stream>>>(Wl2, Wl2T, D1c, D2c);
  wT_bf16<<<grid1(D1c * D2c, 256), 256, 0, stream>>>(Wr2, Wr2T, D1c, D2c);
  wmma_gemm_bf16<<<dim3(D2c / 16, BK1 / 64), 128, 0, stream>>>(xkbf, Wl2T, bl2, xl2,
                                                               BK1, D1c, D2c);
  wmma_gemm_bf16<<<dim3(D2c / 16, BK1 / 64), 128, 0, stream>>>(xkbf, Wr2T, br2, xr2,
                                                               BK1, D1c, D2c);
  // ---- GAT layer 2 attention
  edge_score<<<grid1(M2, 8), 256, 0, stream>>>(xl2, xr2, att2, s1e, d1e, v1e,
                                               e2, max2, K1c, H2c, M2);
  edge_denom<<<grid1(M2, 256), 256, 0, stream>>>(d1e, v1e, max2, e2, den2, K1c, H2c, M2);
  edge_agg<<<grid1(M2, 8), 256, 0, stream>>>(xl2, s1e, d1e, v1e, e2, den2,
                                             agg2, K1c, H2c, M2);
  bn_elu_score<<<grid1(BK1, 8), 256, 0, stream>>>(agg2, bias2, g2, be2, mu2, var2,
                                                  pw2, norms + 1, h2, sc2, D2c, BK1);
  // ---- pool 2 + mean + MLP head
  topk_kernel<<<Bc, 512, 0, stream>>>(sc2, K1c, K2c, idx2, val2, nullptr);
  pool_mean<<<Bc, D2c, 0, stream>>>(h2, idx2, val2, pooled);
  mlp_head<<<1, 256, 0, stream>>>(pooled, W3, b3, W4, b4, (float*)d_out);
}
